// SelfAttentionConv2d_18769007083618
// MI455X (gfx1250) — matrix operand order repaired and checked by reading, not compiled, not run on hardware
//
#include <hip/hip_runtime.h>

typedef __attribute__((ext_vector_type(16))) __bf16 v16bf;
typedef __attribute__((ext_vector_type(8)))  __bf16 v8bf;
typedef __attribute__((ext_vector_type(2)))  __bf16 v2bf;
typedef __attribute__((ext_vector_type(8)))  float  v8f;

#define DQ 32
#define DK 32
#define DV 128
#define B_  16
#define C_  256
#define HW  4096   // 64*64
#define NKV 1024   // 32*32

#define WMMA_BF16(A, Bm, Cacc) \
    __builtin_amdgcn_wmma_f32_16x16x32_bf16(false, (A), false, (Bm), (short)0, (Cacc), false, false)

// A-fragment K index mapping for 16-bit 16x32 A tiles (wave32):
// lane L: M = L%16, kh = L/16; elem e -> K = kh*16+e
__device__ __forceinline__ int amapK(int kh, int e) {
    return kh * 16 + e;
}

// ---------------------------------------------------------------------------
// Pack an f32 weight matrix [M x ld] into bf16 A-fragment order:
// frag[(mt*nks + ks)*512 + lane*16 + e].  One wave per (mt,ks) fragment.
// ---------------------------------------------------------------------------
__global__ __launch_bounds__(32)
void pack_w_kernel(const float* __restrict__ w, __bf16* __restrict__ frag,
                   int nks, int ld) {
    const int mt = blockIdx.x / nks, ks = blockIdx.x % nks;
    const int lane = threadIdx.x, lm = lane & 15, kh = lane >> 4;
    v16bf a;
#pragma unroll
    for (int e = 0; e < 16; ++e)
        a[e] = (__bf16)w[(mt * 16 + lm) * ld + ks * 32 + amapK(kh, e)];
    *(v16bf*)&frag[(size_t)blockIdx.x * 512 + lane * 16] = a;
}

// ---------------------------------------------------------------------------
// Kernel 1: qkv = W(192x256) @ X(256x4096) per batch.
// One wave per (b, ntile) computes ALL 12 M-tiles; W A-fragments are 32B
// loads (L2-hot).  Epilogue transposes each D-tile through LDS so stores
// become contiguous 16B vector stores.
// grid = (256 ntiles, 16 batches), block = 32.
// ---------------------------------------------------------------------------
__global__ __launch_bounds__(32)
void qkv_gemm_kernel(const float* __restrict__ x, const __bf16* __restrict__ wfrag,
                     __bf16* __restrict__ qout, __bf16* __restrict__ kvout) {
    const int nt = blockIdx.x, b = blockIdx.y;
    const int lane = threadIdx.x, lm = lane & 15, kh = lane >> 4;
    const float* xb = x + (size_t)b * C_ * HW;
    const int p = nt * 16 + lm;
    __shared__ __align__(16) __bf16 sT[16][16];   // D-tile transpose buffer

    v8f acc[12] = {};
    for (int ks = 0; ks < 8; ++ks) {
        v16bf bx;
#pragma unroll
        for (int e = 0; e < 16; ++e)
            bx[e] = (__bf16)xb[(size_t)(ks * 32 + kh * 16 + e) * HW + p];
        if (ks < 7)  // CDNA5 speculative prefetch of next k-slab
            __builtin_prefetch(&xb[(size_t)((ks + 1) * 32 + kh * 16) * HW + p], 0, 0);
#pragma unroll
        for (int mt = 0; mt < 12; ++mt) {
            const v16bf aw = *(const v16bf*)&wfrag[(size_t)(mt * 8 + ks) * 512 + lane * 16];
            acc[mt] = WMMA_BF16(aw, bx, acc[mt]);
        }
    }
    // epilogue: LDS transpose -> contiguous 16B row-segment stores
    for (int mt = 0; mt < 12; ++mt) {
#pragma unroll
        for (int r = 0; r < 8; ++r)
            sT[(lane < 16) ? r : (8 + r)][lm] = (__bf16)acc[mt][r];
        __syncthreads();   // single wave: S_NOP; LDS same-wave ops are in-order
        const v8bf row = *(const v8bf*)&sT[lm][kh * 8];
        const int o = mt * 16 + lm;          // o<DQ is uniform per mt
        __bf16* dst = (o < DQ)
            ? &qout[((size_t)b * DQ + o) * HW]
            : &kvout[((size_t)b * (DK + DV) + (o - DQ)) * HW];
        *(v8bf*)&dst[nt * 16 + kh * 8] = row;
        __syncthreads();
    }
}

// ---------------------------------------------------------------------------
// Kernel 2: 2x2 maxpool on k/v.  k is written directly in B-fragment order
// for the attention score WMMAs; v stays row-major (contiguous-K loads).
// ---------------------------------------------------------------------------
__global__ __launch_bounds__(256)
void pool_kernel(const __bf16* __restrict__ kv,
                 __bf16* __restrict__ kfrag, __bf16* __restrict__ vp) {
    int idx = blockIdx.x * 256 + threadIdx.x;
    if (idx >= B_ * (DK + DV) * NKV) return;
    int pp = idx & (NKV - 1);
    int t  = idx >> 10;
    int c  = t % (DK + DV);
    int b  = t / (DK + DV);
    int py = pp >> 5, px = pp & 31;
    const __bf16* src = kv + ((size_t)b * (DK + DV) + c) * HW;
    int p0 = (2 * py) * 64 + 2 * px;
    const v2bf a0 = *(const v2bf*)&src[p0];        // 2 adjacent pixels, b32 load
    const v2bf a1 = *(const v2bf*)&src[p0 + 64];
    float m = fmaxf(fmaxf((float)a0[0], (float)a0[1]),
                    fmaxf((float)a1[0], (float)a1[1]));
    __bf16 v = (__bf16)m;
    if (c < DK) {
        // B-fragment order: kfrag[b][kt][lane = (c/16)*16 + pp%16][e = c%16]
        int kt = pp >> 4, lmn = pp & 15, khh = c >> 4, e = c & 15;
        kfrag[(((size_t)b * 64 + kt) * 32 + (khh * 16 + lmn)) * 16 + e] = v;
    } else {
        vp[((size_t)b * DV + (c - DK)) * NKV + pp] = v;
    }
}

// ---------------------------------------------------------------------------
// Kernel 3: attention, one wave per 16 queries.
//   scores = qT(16x32) @ k(32x1024): 64 WMMAs, k B-frags are single 32B loads
//   softmax: all 32 lanes (2 per row), vectorized 16B LDS scans
//   PV = attn(16x1024) @ vT(1024x128): A-frag from LDS (2x16B), B = 32B load
//   result transposed through LDS into B-fragment order for kernel 4.
// grid = (256 qtiles, 16 batches), block = 32.
// ---------------------------------------------------------------------------
__global__ __launch_bounds__(32)
void attn_kernel(const __bf16* __restrict__ q, const __bf16* __restrict__ kfrag,
                 const __bf16* __restrict__ vp, __bf16* __restrict__ afrag) {
    const int qt = blockIdx.x, b = blockIdx.y;
    const int lane = threadIdx.x, lm = lane & 15, kh = lane >> 4;

    __shared__ __align__(32) unsigned char smem[16 * NKV * 2];   // 32 KB
    __bf16 (*sS)[NKV] = (__bf16 (*)[NKV])smem;   // scores / exp(scores)
    float  (*sO)[16]  = (float  (*)[16]) smem;   // [128 vd][16 p] transpose buf

    const __bf16* qb = q  + (size_t)b * DQ * HW;
    const __bf16* kb = kfrag + (size_t)b * 64 * 512;
    const __bf16* vb = vp + (size_t)b * DV * NKV;

    // q A-fragment (one scattered gather, reused by all 64 key tiles)
    v16bf aq;
    const int qcol = qt * 16 + lm;
#pragma unroll
    for (int e = 0; e < 16; ++e)
        aq[e] = qb[(size_t)amapK(kh, e) * HW + qcol];

    // ---- scores ----
    for (int kt = 0; kt < 64; ++kt) {
        const v16bf bk = *(const v16bf*)&kb[(size_t)kt * 512 + lane * 16];
        v8f acc = {};
        acc = WMMA_BF16(aq, bk, acc);
        int kcol = kt * 16 + lm;
#pragma unroll
        for (int r = 0; r < 8; ++r) {
            int row = (lane < 16) ? r : (8 + r);
            sS[row][kcol] = (__bf16)acc[r];
        }
    }
    __syncthreads();

    // ---- softmax: lane pair (lm, lm+16) owns row lm, halves kh ----
    const int j0 = kh * 512, j1 = j0 + 512;
    float pmax = -3.402823e38f;
    for (int j = j0; j < j1; j += 8) {
        v8bf cv = *(const v8bf*)&sS[lm][j];
#pragma unroll
        for (int t = 0; t < 8; ++t) pmax = fmaxf(pmax, (float)cv[t]);
    }
    const float rmax = fmaxf(pmax, __shfl_xor(pmax, 16, 32));
    float psum = 0.0f;
    for (int j = j0; j < j1; j += 8) {
        v8bf cv = *(const v8bf*)&sS[lm][j];
        v8bf ev;
#pragma unroll
        for (int t = 0; t < 8; ++t) {
            float e = __expf((float)cv[t] - rmax);
            psum += e;
            ev[t] = (__bf16)e;
        }
        *(v8bf*)&sS[lm][j] = ev;
    }
    const float rsum = psum + __shfl_xor(psum, 16, 32);
    __syncthreads();

    // ---- PV ----
    v8f oacc[8] = {};
    for (int ks = 0; ks < 32; ++ks) {
        const int kbase = ks * 32;
        const v8bf lo = *(const v8bf*)&sS[lm][kbase + kh * 16];
        const v8bf hi = *(const v8bf*)&sS[lm][kbase + kh * 16 + 8];
        const v16bf aa = __builtin_shufflevector(lo, hi,
            0, 1, 2, 3, 4, 5, 6, 7, 8, 9, 10, 11, 12, 13, 14, 15);
#pragma unroll
        for (int vt = 0; vt < 8; ++vt) {
            int vd = vt * 16 + lm;
            const v16bf bv = *(const v16bf*)&vb[(size_t)vd * NKV + kbase + kh * 16];
            oacc[vt] = WMMA_BF16(aa, bv, oacc[vt]);
        }
    }
    __syncthreads();   // scores no longer needed; smem becomes sO

    // ---- normalize + transpose into B-fragment order for output GEMM ----
#pragma unroll
    for (int r = 0; r < 8; ++r) {
        int qrow = (lane < 16) ? r : (8 + r);
        float rinv = 1.0f / __shfl(rsum, qrow, 32);
#pragma unroll
        for (int vt = 0; vt < 8; ++vt)
            sO[vt * 16 + lm][qrow] = oacc[vt][r] * rinv;
    }
    __syncthreads();

    __bf16* ab = afrag + (((size_t)(b * 256 + qt)) * 4) * 512;
#pragma unroll
    for (int ks = 0; ks < 4; ++ks) {
        v16bf f;
#pragma unroll
        for (int e = 0; e < 16; ++e)
            f[e] = (__bf16)sO[ks * 32 + kh * 16 + e][lm];
        *(v16bf*)&ab[(size_t)ks * 512 + lane * 16] = f;   // contiguous 32B store
    }
}

// ---------------------------------------------------------------------------
// Kernel 4: final = imgs + scale * (out_w(256x128) @ attn_out(128x4096)).
// One wave per (b, ntile), all 16 M-tiles; A and B frags are 32B loads.
// Epilogue transposes D-tiles through LDS -> contiguous b128 residual
// loads/stores against the two biggest HBM streams.
// grid = (256 ntiles, 16 batches), block = 32.
// ---------------------------------------------------------------------------
__global__ __launch_bounds__(32)
void out_gemm_kernel(const float* __restrict__ imgs, const __bf16* __restrict__ owfrag,
                     const __bf16* __restrict__ afrag, const float* __restrict__ scale,
                     float* __restrict__ out) {
    const int nt = blockIdx.x, b = blockIdx.y;
    const int lane = threadIdx.x, lm = lane & 15, kh = lane >> 4;
    __shared__ __align__(16) float sTf[16][16];

    v8f acc[16] = {};
    for (int ks = 0; ks < 4; ++ks) {
        const v16bf bx = *(const v16bf*)&afrag[(((size_t)(b * 256 + nt)) * 4 + ks) * 512
                                               + lane * 16];
#pragma unroll
        for (int mt = 0; mt < 16; ++mt) {
            const v16bf aw = *(const v16bf*)&owfrag[(size_t)(mt * 4 + ks) * 512 + lane * 16];
            acc[mt] = WMMA_BF16(aw, bx, acc[mt]);
        }
    }
    const float s = scale[0];
    for (int mt = 0; mt < 16; ++mt) {
#pragma unroll
        for (int r = 0; r < 8; ++r)
            sTf[(lane < 16) ? r : (8 + r)][lm] = acc[mt][r];
        __syncthreads();   // single wave: S_NOP; LDS same-wave ops in-order
        const float4 d0 = *(const float4*)&sTf[lm][kh * 8];
        const float4 d1 = *(const float4*)&sTf[lm][kh * 8 + 4];
        const size_t base = ((size_t)b * C_ + mt * 16 + lm) * HW + nt * 16 + kh * 8;
        const float4 i0 = *(const float4*)&imgs[base];
        const float4 i1 = *(const float4*)&imgs[base + 4];
        float4 o0, o1;
        o0.x = i0.x + s * d0.x;  o0.y = i0.y + s * d0.y;
        o0.z = i0.z + s * d0.z;  o0.w = i0.w + s * d0.w;
        o1.x = i1.x + s * d1.x;  o1.y = i1.y + s * d1.y;
        o1.z = i1.z + s * d1.z;  o1.w = i1.w + s * d1.w;
        *(float4*)&out[base]     = o0;
        *(float4*)&out[base + 4] = o1;
        __syncthreads();
    }
}

// ---------------------------------------------------------------------------
extern "C" void kernel_launch(void* const* d_in, const int* in_sizes, int n_in,
                              void* d_out, int out_size, void* d_ws, size_t ws_size,
                              hipStream_t stream) {
    const float* imgs  = (const float*)d_in[0];   // [16,256,64,64]
    const float* qkv_w = (const float*)d_in[1];   // [192,256]
    const float* out_w = (const float*)d_in[2];   // [256,128]
    const float* scale = (const float*)d_in[3];   // [1]
    float* out = (float*)d_out;

    char* ws = (char*)d_ws;
    size_t off = 0;
    __bf16* q_bf   = (__bf16*)(ws + off); off += (size_t)B_ * DQ * HW * 2;          // 4 MB
    __bf16* kv_bf  = (__bf16*)(ws + off); off += (size_t)B_ * (DK + DV) * HW * 2;   // 20 MB
    __bf16* k_frag = (__bf16*)(ws + off); off += (size_t)B_ * 64 * 512 * 2;         // 1 MB
    __bf16* v_pool = (__bf16*)(ws + off); off += (size_t)B_ * DV * NKV * 2;         // 4 MB
    __bf16* a_frag = (__bf16*)(ws + off); off += (size_t)B_ * 256 * 4 * 512 * 2;    // 16 MB
    __bf16* w_frag = (__bf16*)(ws + off); off += (size_t)12 * 8 * 512 * 2;          // 96 KB
    __bf16* ow_frag= (__bf16*)(ws + off); off += (size_t)16 * 4 * 512 * 2;          // 64 KB

    // 0) pack weights into A-fragment layouts
    pack_w_kernel<<<12 * 8, 32, 0, stream>>>(qkv_w, w_frag, 8, C_);
    pack_w_kernel<<<16 * 4, 32, 0, stream>>>(out_w, ow_frag, 4, DV);

    // 1) QKV GEMM
    qkv_gemm_kernel<<<dim3(HW / 16, B_), 32, 0, stream>>>(imgs, w_frag, q_bf, kv_bf);

    // 2) 2x2 maxpool on k/v (k emitted in B-fragment order)
    {
        int n = B_ * (DK + DV) * NKV;
        pool_kernel<<<(n + 255) / 256, 256, 0, stream>>>(kv_bf, k_frag, v_pool);
    }

    // 3) attention
    attn_kernel<<<dim3(HW / 16, B_), 32, 0, stream>>>(q_bf, k_frag, v_pool, a_frag);

    // 4) output projection + residual
    out_gemm_kernel<<<dim3(HW / 16, B_), 32, 0, stream>>>(imgs, ow_frag, a_frag,
                                                          scale, out);
}
